// HTSA_sa_37503654428921
// MI455X (gfx1250) — matrix operand order repaired and checked by reading, not compiled, not run on hardware
//
#include <hip/hip_runtime.h>
#include <math.h>

// ---------------------------------------------------------------------------
// HTSA for MI455X (gfx1250, wave32).
// Final 3D conv = 29 GFLOP vs ~0.5 GB traffic -> memory bound (~20us floor at
// 23.3 TB/s). bf16 WMMA (v_wmma_f32_16x16x32_bf16, f32 accum) keeps the
// matrix pipe off the critical path; htsa is materialized bf16 in an
// i-fastest layout [n][c][h][w][i16] so the conv's LDS tile load is a
// contiguous, layout-preserving copy -> GLOBAL_LOAD_ASYNC_TO_LDS_B128
// (ASYNCcnt) with a vectorized fallback if the builtin is unavailable.
// ---------------------------------------------------------------------------

#define NB 8
#define TB 8
#define CB 128
#define HB 64
#define WB 64
#define HWB 4096
#define KCSEL 64
#define KTSEL 4

typedef __bf16 bf16_t;
typedef __attribute__((ext_vector_type(16))) __bf16 v16bf;
typedef __attribute__((ext_vector_type(8)))  __bf16 v8bf;
typedef __attribute__((ext_vector_type(8)))  float  v8f;
typedef __attribute__((ext_vector_type(4)))  int    v4i;

typedef __attribute__((address_space(1))) v4i g_v4i;
typedef __attribute__((address_space(3))) v4i l_v4i;

union AV { v16bf v; v8bf h[2]; };

#if __has_builtin(__builtin_amdgcn_global_load_async_to_lds_b128)
#define HAVE_ASYNC_LDS 1
#else
#define HAVE_ASYNC_LDS 0
#endif

__device__ __forceinline__ void async_copy16(const bf16_t* g, bf16_t* l) {
#if HAVE_ASYNC_LDS
  __builtin_amdgcn_global_load_async_to_lds_b128((g_v4i*)g, (l_v4i*)l, 0, 0);
#else
  *(uint4*)l = *(const uint4*)g;   // global_load_b128 + ds_store_b128
#endif
}

__device__ __forceinline__ void wait_async_lds() {
#if HAVE_ASYNC_LDS
#if __has_builtin(__builtin_amdgcn_s_wait_asynccnt)
  __builtin_amdgcn_s_wait_asynccnt(0);
#else
  asm volatile("s_wait_asynccnt 0" ::: "memory");
#endif
#endif
}

// --------------------------- K1: M = mean + max over H*W -------------------
__global__ __launch_bounds__(256) void k_meanmax(const float* __restrict__ x,
                                                 float* __restrict__ M) {
  int b = blockIdx.x;                       // (n*T + t)*C + c
  const float* p = x + (size_t)b * HWB;
  float s = 0.f, mx = -3.4e38f;
  for (int i = threadIdx.x; i < HWB; i += 256) {
    float v = p[i];
    s += v;
    mx = fmaxf(mx, v);
  }
  __shared__ float ss[256], sm[256];
  ss[threadIdx.x] = s; sm[threadIdx.x] = mx;
  __syncthreads();
  for (int st = 128; st > 0; st >>= 1) {
    if (threadIdx.x < st) {
      ss[threadIdx.x] += ss[threadIdx.x + st];
      sm[threadIdx.x] = fmaxf(sm[threadIdx.x], sm[threadIdx.x + st]);
    }
    __syncthreads();
  }
  if (threadIdx.x == 0) M[b] = ss[0] / (float)HWB + sm[0];
}

// --------------------------- K2: importance mask (top-k via rank count) ----
__global__ __launch_bounds__(256) void k_imp(const float* __restrict__ M,
                                             float* __restrict__ imp) {
  int idx = blockIdx.x * 256 + threadIdx.x;  // t*C + c
  if (idx >= TB * CB) return;
  int t = idx / CB, c = idx % CB;
  int mch = 0, mtm = 0;
  for (int n = 0; n < NB; ++n) {
    const float* Mn = M + n * TB * CB;
    float v = Mn[t * CB + c];
    int rc = 0;
    for (int cc = 0; cc < CB; ++cc) rc += (Mn[t * CB + cc] > v) ? 1 : 0;
    if (rc < KCSEL) mch = 1;
    int rt = 0;
    for (int tt = 0; tt < TB; ++tt) rt += (Mn[tt * CB + c] > v) ? 1 : 0;
    if (rt < KTSEL) mtm = 1;
  }
  imp[idx] = ((mch + mtm) == 1) ? 1.0f : 0.0f;
}

// --------------------------- K3: channel stats (avg/max of im & sub) -------
__global__ __launch_bounds__(256) void k_stats(const float* __restrict__ x,
                                               const float* __restrict__ imp,
                                               float* __restrict__ stats) {
  int gid = blockIdx.x * 256 + threadIdx.x;  // over N*T*HW (HW % 256 == 0)
  int nt = gid / HWB, hw = gid % HWB;
  int t = nt % TB;
  __shared__ float simp[CB];
  if (threadIdx.x < CB) simp[threadIdx.x] = imp[t * CB + threadIdx.x];
  __syncthreads();
  const float* p = x + (size_t)nt * CB * HWB + hw;
  float s_im = 0.f, s_sub = 0.f, m_im = -3.4e38f, m_sub = -3.4e38f;
  for (int c = 0; c < CB; ++c) {
    float v = p[(size_t)c * HWB];
    float w = simp[c];
    float vi = w * v;
    float vs = v - vi;
    s_im += vi; s_sub += vs;
    m_im = fmaxf(m_im, vi); m_sub = fmaxf(m_sub, vs);
  }
  const int plane = NB * TB * HWB;
  stats[gid]             = s_im  * (1.0f / 64.0f);   // mean/LAM  = sum/(C*0.5)
  stats[plane + gid]     = m_im;
  stats[2 * plane + gid] = s_sub * (1.0f / 64.0f);   // mean/(1-LAM)
  stats[3 * plane + gid] = m_sub;
}

// --------------------------- K4: 3x3 2-ch conv + sigmoid -> maps -----------
__global__ __launch_bounds__(256) void k_maps(const float* __restrict__ stats,
                                              const float* __restrict__ w1,
                                              const float* __restrict__ w2,
                                              float* __restrict__ im_map,
                                              float* __restrict__ sub_map) {
  int gid = blockIdx.x * 256 + threadIdx.x;  // N*T*HW
  int nt = gid / HWB, hw = gid % HWB;
  int h = hw / WB, w = hw % WB;
  const int plane = NB * TB * HWB;
  const float* base = stats + (size_t)nt * HWB;
  float acc1 = 0.f, acc2 = 0.f;
  for (int kh = 0; kh < 3; ++kh) {
    int hh = h + kh - 1;
    if (hh < 0 || hh >= HB) continue;
    for (int kw = 0; kw < 3; ++kw) {
      int ww = w + kw - 1;
      if (ww < 0 || ww >= WB) continue;
      int off = hh * WB + ww;
      acc1 += base[off]             * w1[(0 * 3 + kh) * 3 + kw]
            + base[plane + off]     * w1[(1 * 3 + kh) * 3 + kw];
      acc2 += base[2 * plane + off] * w2[(0 * 3 + kh) * 3 + kw]
            + base[3 * plane + off] * w2[(1 * 3 + kh) * 3 + kw];
    }
  }
  im_map[gid]  = 1.f / (1.f + expf(-acc1));
  sub_map[gid] = 1.f / (1.f + expf(-acc2));
}

// --------------------------- K5: htsa (bf16), i-fastest layout --------------
// htsa2[n][c][h][w][i16]: one thread per (n,c,h,w) computes all 16 conv input
// channels (im branch t, sub branch t+8) and stores one 32B chunk (coalesced).
__global__ __launch_bounds__(256) void k_htsa2(const float* __restrict__ x,
                                               const float* __restrict__ imp,
                                               const float* __restrict__ im_map,
                                               const float* __restrict__ sub_map,
                                               bf16_t* __restrict__ htsa2) {
  int gid = blockIdx.x * 256 + threadIdx.x;  // (n*CB + c)*HWB + hw
  int hw = gid % HWB;
  int nc = gid / HWB;
  int c = nc % CB, n = nc / CB;
  union { bf16_t e[16]; uint4 q[2]; } v;
#pragma unroll
  for (int t = 0; t < TB; ++t) {
    float xv  = x[((size_t)(n * TB + t) * CB + c) * HWB + hw];
    float ip  = imp[t * CB + c];
    size_t nt = (size_t)(n * TB + t) * HWB + hw;
    float imv = im_map[nt];
    float sbv = sub_map[nt];
    v.e[t]     = (bf16_t)(ip * imv * xv);
    v.e[t + 8] = (bf16_t)((1.f - ip) * sbv * xv);
  }
  uint4* dst = (uint4*)(htsa2 + (size_t)gid * 16);
  dst[0] = v.q[0];
  dst[1] = v.q[1];
}

// --------------------------- K6: pack conv3d weights to bf16 B layout ------
// K ordering: k = q*16 + i, q = (kd*3+kh)*3+kw (27), i = in-channel (16);
// padded to 14 K-blocks of 32. Bw[kb][col16][kloc32].
__global__ void k_wb(const float* __restrict__ wl, bf16_t* __restrict__ Bw) {
  for (int idx = threadIdx.x; idx < 14 * 16 * 32; idx += blockDim.x) {
    int kloc = idx % 32;
    int col  = (idx / 32) % 16;
    int kb   = idx / (32 * 16);
    int k = kb * 32 + kloc;
    int q = k >> 4, i = k & 15;
    float v = 0.f;
    if (q < 27 && col < 8) {
      int kd = q / 9, kh = (q % 9) / 3, kw = q % 3;
      v = wl[(((col * 16 + i) * 3 + kd) * 3 + kh) * 3 + kw];
    }
    Bw[idx] = (bf16_t)v;
  }
}

// --------------------------- K7: conv3d via bf16 WMMA ----------------------
// Block = (n, d, h-pair). 8 waves = 2 h-rows x 4 w-tiles of 16.
// LDS A tile [dd3][hh4][ww66][i16]: per-lane A octet = fixed q, 8 consecutive
// i -> ds_load_b128. Tile rows are contiguous 2KB runs of htsa2 -> async
// global->LDS B128 copies; halo cells zero-filled at disjoint addresses.
__global__ __launch_bounds__(256) void k_conv3d(const bf16_t* __restrict__ htsa2,
                                                const bf16_t* __restrict__ BwG,
                                                float* __restrict__ out) {
  int bid = blockIdx.x;
  int hp = bid & 31;
  int nd = bid >> 5;
  int d = nd & 127;
  int n = nd >> 7;
  int h0 = hp * 2;

  __shared__ __align__(16) bf16_t sA[3 * 4 * 66 * 16];  // 25344 B
  __shared__ __align__(16) bf16_t sB[14 * 16 * 32];     // 14336 B

  // 1) zero-fill halo: ww==0, ww==65, and rows with out-of-range c'/h'
  //    (disjoint from async destinations -> no ordering hazard)
  const uint4 z4 = make_uint4(0u, 0u, 0u, 0u);
  for (int idx = threadIdx.x; idx < 12 * 132; idx += 256) {  // 8-elem chunks
    int row = idx / 132;               // (dd*4 + hh)
    int c8  = idx % 132;               // chunk within row (1056 elems / 8)
    int dd = row >> 2, hh = row & 3;
    int cc = d + dd - 1, hy = h0 + hh - 1;
    bool rowValid = (cc >= 0 && cc < CB && hy >= 0 && hy < HB);
    int ww = c8 >> 1;                  // 2 chunks per ww point (16 bf16)
    if (!rowValid || ww == 0 || ww == 65)
      *(uint4*)(&sA[row * (66 * 16) + c8 * 8]) = z4;
  }

  // 2) async-copy interior rows (64 w-points * 16 i = 2KB contiguous each)
  for (int idx = threadIdx.x; idx < 12 * 128; idx += 256) {
    int row = idx >> 7;                // 0..11
    int seg = idx & 127;               // 16B segment within row
    int dd = row >> 2, hh = row & 3;
    int cc = d + dd - 1, hy = h0 + hh - 1;
    if (cc < 0 || cc >= CB || hy < 0 || hy >= HB) continue;
    const bf16_t* g = htsa2 + ((size_t)(n * CB + cc) * HWB + hy * WB) * 16 + seg * 8;
    bf16_t* l = &sA[(row * 66 + 1) * 16 + seg * 8];
    async_copy16(g, l);
  }
  // 3) weights -> LDS (contiguous both sides)
  for (int idx = threadIdx.x; idx < (14 * 16 * 32) / 8; idx += 256)
    async_copy16(BwG + idx * 8, &sB[idx * 8]);

  wait_async_lds();
  __syncthreads();

  int lane = threadIdx.x & 31;
  int wv   = threadIdx.x >> 5;
  int r    = wv >> 2;        // output row within h-pair
  int wt   = wv & 3;         // w tile (16 wide)
  int m    = lane & 15;      // A: M row / B: column / D: N column
  int s    = lane >> 4;      // half-wave select

  v8f acc;
#pragma unroll
  for (int j = 0; j < 8; ++j) acc[j] = 0.f;

  v8bf zero8;
#pragma unroll
  for (int j = 0; j < 8; ++j) zero8[j] = (bf16_t)0.f;

#pragma unroll
  for (int kb = 0; kb < 14; ++kb) {
    const int q0 = 2 * kb, q1 = 2 * kb + 1;
    AV a, b;
    {
      int kd = q0 / 9, kh = (q0 % 9) / 3, kw = q0 % 3;
      int off = (((kd * 4 + (kh + r)) * 66) + (wt * 16 + m + kw)) * 16 + 8 * s;
      a.h[0] = *(const v8bf*)(&sA[off]);
    }
    if (q1 < 27) {
      int kd = q1 / 9, kh = (q1 % 9) / 3, kw = q1 % 3;
      int off = (((kd * 4 + (kh + r)) * 66) + (wt * 16 + m + kw)) * 16 + 8 * s;
      a.h[1] = *(const v8bf*)(&sA[off]);
    } else {
      a.h[1] = zero8;
    }
    {
      int offB = (kb * 16 + m) * 32 + s * 8;
      b.h[0] = *(const v8bf*)(&sB[offB]);
      b.h[1] = *(const v8bf*)(&sB[offB + 16]);
    }
    acc = __builtin_amdgcn_wmma_f32_16x16x32_bf16(
        /*neg_a=*/false, a.v, /*neg_b=*/false, b.v,
        /*c_mod=*/(short)0, acc, /*reuse_a=*/false, /*reuse_b=*/false);
  }

  // D layout: lane holds N=col (lane&15), M = v + 8*(lane>>4)
  if (m < 8) {
    int o = m;
    int h = h0 + r;
    size_t obase = (((size_t)(n * 8 + o) * CB + d) * HWB) + (size_t)h * WB
                 + wt * 16 + 8 * s;
    *(float4*)(&out[obase])     = make_float4(acc[0], acc[1], acc[2], acc[3]);
    *(float4*)(&out[obase + 4]) = make_float4(acc[4], acc[5], acc[6], acc[7]);
  }
}

// ---------------------------------------------------------------------------
extern "C" void kernel_launch(void* const* d_in, const int* in_sizes, int n_in,
                              void* d_out, int out_size, void* d_ws, size_t ws_size,
                              hipStream_t stream) {
  const float* x  = (const float*)d_in[0];
  const float* w1 = (const float*)d_in[1];
  const float* w2 = (const float*)d_in[2];
  const float* wl = (const float*)d_in[3];
  float* out = (float*)d_out;

  float* M       = (float*)d_ws;                         // 8192
  float* imp     = M + NB * TB * CB;                     // 1024
  float* stats   = imp + TB * CB;                        // 4 * 262144
  float* im_map  = stats + 4 * NB * TB * HWB;            // 262144
  float* sub_map = im_map + NB * TB * HWB;               // 262144
  bf16_t* Bw     = (bf16_t*)(sub_map + NB * TB * HWB);   // 7168 bf16 (14336 B)
  bf16_t* htsa2  = (bf16_t*)((char*)Bw + 14 * 16 * 32 * sizeof(bf16_t));

  k_meanmax<<<NB * TB * CB, 256, 0, stream>>>(x, M);
  k_imp<<<(TB * CB + 255) / 256, 256, 0, stream>>>(M, imp);
  k_stats<<<NB * TB * HWB / 256, 256, 0, stream>>>(x, imp, stats);
  k_maps<<<NB * TB * HWB / 256, 256, 0, stream>>>(stats, w1, w2, im_map, sub_map);
  k_htsa2<<<(NB * CB * HWB) / 256, 256, 0, stream>>>(x, imp, im_map, sub_map, htsa2);
  k_wb<<<1, 256, 0, stream>>>(wl, Bw);
  k_conv3d<<<NB * CB * (HB / 2), 256, 0, stream>>>(htsa2, Bw, out);
}
